// MultiScaleRetention_63848983822574
// MI455X (gfx1250) — matrix operand order repaired and checked by reading, not compiled, not run on hardware
//
#include <hip/hip_runtime.h>
#include <cstdint>

#define B_    2
#define T_    2048
#define H_    6
#define DQK_  128
#define DV_   256
#define DM_   768
#define HDV_  1536
#define MTOT_ (B_*T_)   // 4096

typedef __attribute__((ext_vector_type(16))) _Float16 v16h;
typedef __attribute__((ext_vector_type(8)))  float    v8f;
typedef __attribute__((ext_vector_type(4)))  uint32_t u32x4;

union AF {
  v16h  v;
  uint32_t u[8];
  u32x4 q[2];
};

static __device__ __forceinline__ v8f wmma_f16(v16h a, v16h b, v8f c) {
  // (neg_a, A, neg_b, B, c_mod, C, reuse_a, reuse_b)
  return __builtin_amdgcn_wmma_f32_16x16x32_f16(false, a, false, b, (short)0, c, false, false);
}

// A-matrix fragment load (16x32 f16 tile, row-major source, row = lane%16)
static __device__ __forceinline__ void loadA(AF& f, const _Float16* row, int lhi) {
  f.q[0] = *(const u32x4*)(row + 8 * lhi);        // K 0..7  | 8..15
  f.q[1] = *(const u32x4*)(row + 16 + 8 * lhi);   // K 16..23| 24..31
}
// B-matrix fragment load (32x16 f16 tile; source is (col,K) row-major, col = lane%16)
static __device__ __forceinline__ void loadB(AF& f, const _Float16* row, int lhi) {
  f.q[0] = *(const u32x4*)(row + 8 * lhi);        // K 0..7  | 8..15
  f.q[1] = *(const u32x4*)(row + 16 + 8 * lhi);   // K 16..23| 24..31
}

// ---------------- elementwise prep kernels ----------------

__global__ void cvt_f16_kernel(const float* __restrict__ s, _Float16* __restrict__ d, int n) {
  int i = blockIdx.x * 256 + threadIdx.x;
  if (i < n) d[i] = (_Float16)s[i];
}

// dst[n*K + k] = src[k*N + n]   (W is K x N row-major; dst is N x K "Bt" layout)
__global__ void tr_f16_kernel(const float* __restrict__ s, _Float16* __restrict__ d, int K, int N) {
  int i = blockIdx.x * 256 + threadIdx.x;
  if (i >= K * N) return;
  int k = i / N, n = i % N;
  d[(size_t)n * K + k] = (_Float16)s[i];
}

// xPos rotary: src f32 (B,T,H,DQK) -> dst f16 same layout. sign=+1 (Q) / -1 (K)
__global__ void rotary_kernel(const float* __restrict__ s, _Float16* __restrict__ d, float sign) {
  int idx = blockIdx.x * 256 + threadIdx.x;
  const int tot = B_ * T_ * H_ * (DQK_ / 2);
  if (idx >= tot) return;
  int i = idx & 63;
  int h = (idx >> 6) % H_;
  int t = (idx / (64 * H_)) % T_;
  int b = idx / (64 * H_ * T_);
  float fi   = (float)i;
  float invf = __powf(10000.f, -fi / 64.f);
  float ang  = (float)t * invf;
  float zeta = (fi + 64.f) / 128.f;
  float dec  = __powf(zeta, sign * (float)t / 512.f);
  float ca = __cosf(ang) * dec, sa = __sinf(ang) * dec;
  size_t base = ((size_t)((b * T_ + t) * H_ + h)) * DQK_ + i;
  float x1 = s[base], x2 = s[base + 64];
  d[base]      = (_Float16)(x1 * ca - x2 * sa);
  d[base + 64] = (_Float16)(x1 * sa + x2 * ca);
}

// Vf f32 (B*T, H*DV) -> Vt f16 (B,H,DV,T)
__global__ void vtrans_kernel(const float* __restrict__ v, _Float16* __restrict__ vt) {
  int idx = blockIdx.x * 256 + threadIdx.x;
  if (idx >= MTOT_ * HDV_) return;
  int dv = idx % DV_;
  int h  = (idx / DV_) % H_;
  int t  = (idx / HDV_) % T_;
  int b  = idx / (HDV_ * T_);
  vt[((size_t)(b * H_ + h) * DV_ + dv) * T_ + t] = (_Float16)v[idx];
}

// ---------------- WMMA GEMM: C(M,N) = A16(M,K) @ Bt16(N,K)^T ----------------
// Each wave: 16x64 strip (1 A-frag x 4 B-frags), software double-buffered over K.
__global__ __launch_bounds__(128) void gemm16x64_kernel(const _Float16* __restrict__ A,
                                                        const _Float16* __restrict__ Bt,
                                                        float* __restrict__ C,
                                                        int M, int N, int K) {
  const int Ntb = N >> 6;             // 64-wide n strips
  const int Mt  = M >> 4;
  int wid = blockIdx.x * 4 + (threadIdx.x >> 5);
  if (wid >= Mt * Ntb) return;
  const int m0 = (wid / Ntb) << 4, n0 = (wid % Ntb) << 6;
  const int lane = threadIdx.x & 31, l15 = lane & 15, lhi = lane >> 4;
  const _Float16* arow = A  + (size_t)(m0 + l15) * K;
  const _Float16* brow = Bt + (size_t)(n0 + l15) * K;
  v8f zero = {};
  v8f acc[4];
#pragma unroll
  for (int nb = 0; nb < 4; ++nb) acc[nb] = zero;

  AF af, bf[4];
  loadA(af, arow, lhi);
#pragma unroll
  for (int nb = 0; nb < 4; ++nb) loadB(bf[nb], brow + (size_t)nb * 16 * K, lhi);

  for (int kk = 32; kk < K; kk += 32) {
    AF naf, nbf[4];
    loadA(naf, arow + kk, lhi);
#pragma unroll
    for (int nb = 0; nb < 4; ++nb) loadB(nbf[nb], brow + (size_t)nb * 16 * K + kk, lhi);
#pragma unroll
    for (int nb = 0; nb < 4; ++nb) acc[nb] = wmma_f16(af.v, bf[nb].v, acc[nb]);
    af = naf;
#pragma unroll
    for (int nb = 0; nb < 4; ++nb) bf[nb] = nbf[nb];
  }
#pragma unroll
  for (int nb = 0; nb < 4; ++nb) acc[nb] = wmma_f16(af.v, bf[nb].v, acc[nb]);

  float* c0 = C + (size_t)(m0 + 8 * lhi) * N + n0 + l15;
#pragma unroll
  for (int nb = 0; nb < 4; ++nb)
#pragma unroll
    for (int r = 0; r < 8; ++r) c0[(size_t)r * N + nb * 16] = acc[nb][r];
}

// ---------------- retention core ----------------
// One wave per (b, h, 16-query tile). Q,K f16 (B,T,H,DQK); Vt f16 (B,H,DV,T);
// O f32 (B,H,T,DV).
__global__ __launch_bounds__(128) void attn_kernel(const _Float16* __restrict__ Q,
                                                   const _Float16* __restrict__ Kt,
                                                   const _Float16* __restrict__ Vt,
                                                   float* __restrict__ O) {
  const int Tt = T_ / 16;
  int wid = blockIdx.x * 4 + (threadIdx.x >> 5);
  if (wid >= B_ * H_ * Tt) return;
  const int it = wid % Tt;
  const int h  = (wid / Tt) % H_;
  const int b  = wid / (Tt * H_);
  const int lane = threadIdx.x & 31, l15 = lane & 15, lhi = lane >> 4;
  const int wv = threadIdx.x >> 5;
  __shared__ __align__(16) _Float16 sS[4][16 * 32];
  const int i0 = it << 4;

  // resident Q fragments for the 16x128 query tile (4 x 16x32 A-frags)
  AF qf[4];
  const _Float16* qrow = Q + ((size_t)((b * T_ + i0 + l15) * H_ + h)) * DQK_;
#pragma unroll
  for (int c = 0; c < 4; ++c) loadA(qf[c], qrow + 32 * c, lhi);

  v8f zero = {};
  v8f acc[16];
#pragma unroll
  for (int nt = 0; nt < 16; ++nt) acc[nt] = zero;

  // decay parameters: gamma = 1 - 2^-(5+h); rowsum_i = (1 - gamma^{i+1})/(1-gamma)
  const float log2g  = log2f(1.f - exp2f(-(float)(5 + h)));
  const float inv1mg = exp2f((float)(5 + h));
  const float rsd    = 0.088388347648318447f;   // 1/sqrt(128)

  const int nblk = it / 2 + 1;                  // 32-key blocks covering keys <= i0+15
  for (int blk = 0; blk < nblk; ++blk) {
    const int j0 = blk << 5;
#pragma unroll
    for (int s = 0; s < 2; ++s) {
      const int jt = j0 + (s << 4);
      v8f sc = zero;
      const _Float16* krow = Kt + ((size_t)((b * T_ + jt + l15) * H_ + h)) * DQK_;
      AF kf;
      loadB(kf, krow, lhi);
#pragma unroll
      for (int c = 0; c < 4; ++c) {             // B-frag: k-dim = dqk, col = key
        AF nkf;
        if (c < 3) loadB(nkf, krow + 32 * (c + 1), lhi);
        sc = wmma_f16(qf[c].v, kf.v, sc);
        if (c < 3) kf = nkf;
      }
      // scale by 1/sqrt(DQK) and normalized causal decay; park in LDS as f16 A-matrix
      const int j = jt + l15;
#pragma unroll
      for (int r = 0; r < 8; ++r) {
        const int i = i0 + r + 8 * lhi;
        float val = 0.f;
        if (j <= i) {
          float rowsum = (1.f - exp2f((float)(i + 1) * log2g)) * inv1mg;
          val = sc[r] * rsd * exp2f((float)(i - j) * log2g) * rsqrtf(fmaxf(rowsum, 1e-6f));
        }
        sS[wv][(r + 8 * lhi) * 32 + (s << 4) + l15] = (_Float16)val;
      }
    }
    asm volatile("s_wait_dscnt 0" ::: "memory");
    AF af;   // S' (16 queries x 32 keys) as A-frag
    af.q[0] = *(const u32x4*)&sS[wv][l15 * 32 + 8 * lhi];
    af.q[1] = *(const u32x4*)&sS[wv][l15 * 32 + 16 + 8 * lhi];
    const _Float16* vrow = Vt + ((size_t)((b * H_ + h) * DV_ + l15)) * T_ + j0;
    AF bf;
    loadB(bf, vrow, lhi);
#pragma unroll
    for (int nt = 0; nt < 16; ++nt) {           // DV = 256 -> 16 N-tiles
      AF nbf;
      if (nt < 15) loadB(nbf, vrow + (size_t)(nt + 1) * 16 * T_, lhi);
      acc[nt] = wmma_f16(af.v, bf.v, acc[nt]);
      if (nt < 15) bf = nbf;
    }
  }
  float* ob = O + ((size_t)((b * H_ + h) * T_) + i0 + 8 * lhi) * DV_ + l15;
#pragma unroll
  for (int nt = 0; nt < 16; ++nt)
#pragma unroll
    for (int r = 0; r < 8; ++r)
      ob[(size_t)r * DV_ + nt * 16] = acc[nt][r];
}

// ---------------- group norm (stats over T*DV per (b,h)) + gate ----------------

__global__ __launch_bounds__(256) void stats_kernel(const float* __restrict__ O,
                                                    float* __restrict__ st) {
  const int bh = blockIdx.x;
  const int n = T_ * DV_;  // 524288
  const float* p = O + (size_t)bh * n;
  float s = 0.f, s2 = 0.f;
  for (int i = threadIdx.x; i < n; i += 256) { float v = p[i]; s += v; s2 += v * v; }
  __shared__ float sh[256], sh2[256];
  sh[threadIdx.x] = s; sh2[threadIdx.x] = s2;
  __syncthreads();
  for (int ofs = 128; ofs > 0; ofs >>= 1) {
    if ((int)threadIdx.x < ofs) {
      sh[threadIdx.x]  += sh[threadIdx.x + ofs];
      sh2[threadIdx.x] += sh2[threadIdx.x + ofs];
    }
    __syncthreads();
  }
  if (threadIdx.x == 0) {
    float mean = sh[0] / (float)n;
    float var  = sh2[0] / (float)n - mean * mean;
    st[2 * bh]     = mean;
    st[2 * bh + 1] = rsqrtf(var + 1e-5f);
  }
}

__global__ void gatenorm_kernel(const float* __restrict__ O, const float* __restrict__ st,
                                const float* __restrict__ G, const float* __restrict__ gw,
                                const float* __restrict__ gb, _Float16* __restrict__ A2) {
  int idx = blockIdx.x * 256 + threadIdx.x;
  if (idx >= MTOT_ * HDV_) return;
  int c = idx % HDV_;
  int m = idx / HDV_;           // m = b*T + t
  int h = c / DV_, dv = c % DV_;
  int b = m / T_, t = m % T_;
  int bh = b * H_ + h;
  float mean = st[2 * bh], rstd = st[2 * bh + 1];
  float o  = O[((size_t)bh * T_ + t) * DV_ + dv];
  float nv = (o - mean) * rstd * gw[c] + gb[c];
  float g  = G[idx];
  float gate = g / (1.f + __expf(-g));          // silu
  A2[idx] = (_Float16)(nv * gate);
}

// ---------------- launch ----------------

extern "C" void kernel_launch(void* const* d_in, const int* in_sizes, int n_in,
                              void* d_out, int out_size, void* d_ws, size_t ws_size,
                              hipStream_t stream) {
  (void)in_sizes; (void)n_in; (void)out_size; (void)ws_size;
  const float* x  = (const float*)d_in[0];
  const float* Wq = (const float*)d_in[1];
  const float* Wk = (const float*)d_in[2];
  const float* Wv = (const float*)d_in[3];
  const float* Wg = (const float*)d_in[4];
  const float* Wo = (const float*)d_in[5];
  const float* gw = (const float*)d_in[6];
  const float* gb = (const float*)d_in[7];
  float* out = (float*)d_out;
  char* ws = (char*)d_ws;

  size_t p = 0;
  _Float16* X16 = (_Float16*)(ws + p); p += (size_t)MTOT_ * DM_ * 2;
  _Float16* WqT = (_Float16*)(ws + p); p += (size_t)DM_ * DM_ * 2;
  _Float16* WkT = (_Float16*)(ws + p); p += (size_t)DM_ * DM_ * 2;
  _Float16* WvT = (_Float16*)(ws + p); p += (size_t)DM_ * HDV_ * 2;
  _Float16* WgT = (_Float16*)(ws + p); p += (size_t)DM_ * HDV_ * 2;
  _Float16* WoT = (_Float16*)(ws + p); p += (size_t)DM_ * HDV_ * 2;
  _Float16* Q16 = (_Float16*)(ws + p); p += (size_t)MTOT_ * DM_ * 2;
  _Float16* K16 = (_Float16*)(ws + p); p += (size_t)MTOT_ * DM_ * 2;
  float*    Gf  = (float*)(ws + p);    p += (size_t)MTOT_ * HDV_ * 4;
  float*    Qf  = (float*)(ws + p);                       // f32 proj out; dead after rotary
  _Float16* A2  = (_Float16*)Qf;       p += (size_t)MTOT_ * DM_ * 4;  // alias: f16 4096x1536
  float*    Kf  = (float*)(ws + p);                       // dead after rotary
  _Float16* Vt16= (_Float16*)Kf;       p += (size_t)MTOT_ * DM_ * 4;  // alias: f16 (B,H,DV,T)
  float*    Vf  = (float*)(ws + p);                       // dead after vtrans
  float*    O   = Vf;                  p += (size_t)MTOT_ * HDV_ * 4; // alias: f32 (B,H,T,DV)
  float*    st  = (float*)(ws + p);    p += 256;

  int n;
  n = MTOT_ * DM_;
  cvt_f16_kernel<<<(n + 255) / 256, 256, 0, stream>>>(x, X16, n);
  n = DM_ * DM_;
  tr_f16_kernel<<<(n + 255) / 256, 256, 0, stream>>>(Wq, WqT, DM_, DM_);
  tr_f16_kernel<<<(n + 255) / 256, 256, 0, stream>>>(Wk, WkT, DM_, DM_);
  n = DM_ * HDV_;
  tr_f16_kernel<<<(n + 255) / 256, 256, 0, stream>>>(Wv, WvT, DM_, HDV_);
  tr_f16_kernel<<<(n + 255) / 256, 256, 0, stream>>>(Wg, WgT, DM_, HDV_);
  tr_f16_kernel<<<(n + 255) / 256, 256, 0, stream>>>(Wo, WoT, HDV_, DM_);

  int tiles = (MTOT_ / 16) * (DM_ / 64);
  int blk1 = (tiles + 3) / 4;
  gemm16x64_kernel<<<blk1, 128, 0, stream>>>(X16, WqT, Qf, MTOT_, DM_, DM_);
  gemm16x64_kernel<<<blk1, 128, 0, stream>>>(X16, WkT, Kf, MTOT_, DM_, DM_);
  tiles = (MTOT_ / 16) * (HDV_ / 64);
  int blk2 = (tiles + 3) / 4;
  gemm16x64_kernel<<<blk2, 128, 0, stream>>>(X16, WvT, Vf, MTOT_, HDV_, DM_);
  gemm16x64_kernel<<<blk2, 128, 0, stream>>>(X16, WgT, Gf, MTOT_, HDV_, DM_);

  n = B_ * T_ * H_ * (DQK_ / 2);
  rotary_kernel<<<(n + 255) / 256, 256, 0, stream>>>(Qf, Q16, 1.f);
  rotary_kernel<<<(n + 255) / 256, 256, 0, stream>>>(Kf, K16, -1.f);

  n = MTOT_ * HDV_;
  vtrans_kernel<<<(n + 255) / 256, 256, 0, stream>>>(Vf, Vt16);

  int waves = B_ * H_ * (T_ / 16);
  attn_kernel<<<(waves + 3) / 4, 128, 0, stream>>>(Q16, K16, Vt16, O);

  stats_kernel<<<B_ * H_, 256, 0, stream>>>(O, st);

  n = MTOT_ * HDV_;
  gatenorm_kernel<<<(n + 255) / 256, 256, 0, stream>>>(O, st, Gf, gw, gb, A2);

  tiles = (MTOT_ / 16) * (DM_ / 64);
  gemm16x64_kernel<<<(tiles + 3) / 4, 128, 0, stream>>>(A2, WoT, out, MTOT_, DM_, HDV_);
}
